// ProbSparseSelfAttention_16999480558139
// MI455X (gfx1250) — compile-verified, matrix operand order checked by reading, not run on hardware
//
#include <hip/hip_runtime.h>
#include <math.h>

typedef __attribute__((ext_vector_type(16))) _Float16 v16h;
typedef __attribute__((ext_vector_type(2)))  _Float16 h2;
typedef __attribute__((ext_vector_type(8)))  float    v8f;

#define S_LEN 2048
#define D_DIM 64
#define KT    64          // keys per LDS tile
#define QT    16          // query rows per M-tile
#define MT    2           // M-tiles per wave -> 32 rows/wave
#define WAVES 8
#define LDK   68          // padded LDS row stride (elements)

// Entire softmax scale folded into Q: scores come out of WMMA in exp2 domain.
#define C2 (0.125f * 1.4426950408889634f)

// ---- butterfly-xor within each 16-lane half via v_permlane16_b32 (VALU, no LDS) ----
#if __has_builtin(__builtin_amdgcn_permlane16)
__device__ __forceinline__ float permx(float v, unsigned s0, unsigned s1) {
  int x = __float_as_int(v);
  int r = __builtin_amdgcn_permlane16(x, x, (int)s0, (int)s1, false, false);
  return __int_as_float(r);
}
__device__ __forceinline__ float red_max16(float x) {
  x = fmaxf(x, permx(x, 0x67452301u, 0xEFCDAB89u));   // xor 1
  x = fmaxf(x, permx(x, 0x54761032u, 0xDCFE98BAu));   // xor 2
  x = fmaxf(x, permx(x, 0x32107654u, 0xBA98FEDCu));   // xor 4
  x = fmaxf(x, permx(x, 0xFEDCBA98u, 0x76543210u));   // xor 8
  return x;
}
__device__ __forceinline__ float red_sum16(float x) {
  x += permx(x, 0x67452301u, 0xEFCDAB89u);
  x += permx(x, 0x54761032u, 0xDCFE98BAu);
  x += permx(x, 0x32107654u, 0xBA98FEDCu);
  x += permx(x, 0xFEDCBA98u, 0x76543210u);
  return x;
}
#else
__device__ __forceinline__ float red_max16(float x) {
  for (int off = 1; off < 16; off <<= 1) x = fmaxf(x, __shfl_xor(x, off, 32));
  return x;
}
__device__ __forceinline__ float red_sum16(float x) {
  for (int off = 1; off < 16; off <<= 1) x += __shfl_xor(x, off, 32);
  return x;
}
#endif

__global__ __launch_bounds__(256)
__attribute__((amdgpu_waves_per_eu(4)))
void flash_attn_f16_wmma(
    const float* __restrict__ Q, const float* __restrict__ K,
    const float* __restrict__ V, float* __restrict__ O)
{
  __shared__ _Float16 sK [KT * LDK];              // [key][d]
  __shared__ _Float16 sVt[D_DIM * LDK];           // [d][key] (V transposed)
  __shared__ _Float16 sP [WAVES][MT * QT * LDK];  // per-wave P scratch [row][key]

  const int tid  = threadIdx.x;
  const int lane = tid & 31;
  const int wave = tid >> 5;
  const int half = lane >> 4;
  const int lc   = lane & 15;

  const int bh    = blockIdx.y;                               // 0..B*H-1
  const int qrow0 = blockIdx.x * (WAVES * MT * QT) + wave * (MT * QT);

  const size_t baseQ  = ((size_t)bh * S_LEN + qrow0) * D_DIM;
  const size_t baseKV = (size_t)bh * S_LEN * D_DIM;

  // ---- Q A-fragments, pre-scaled by C2, f32 -> f16 ----
  v16h aq[MT][2];
  #pragma unroll
  for (int mt = 0; mt < MT; ++mt) {
    const float* qrow = Q + baseQ + (size_t)(mt * QT + lc) * D_DIM;  // A row = lane&15
    #pragma unroll
    for (int v = 0; v < 8; ++v) {
      const int d0 = (v & 3) * 2 + half * 8 + (v >> 2) * 16;
      aq[mt][0][2*v]   = (_Float16)(qrow[d0]          * C2);
      aq[mt][0][2*v+1] = (_Float16)(qrow[d0 + 1]      * C2);
      aq[mt][1][2*v]   = (_Float16)(qrow[32 + d0]     * C2);
      aq[mt][1][2*v+1] = (_Float16)(qrow[32 + d0 + 1] * C2);
    }
  }

  v8f acc[MT][4];
  #pragma unroll
  for (int mt = 0; mt < MT; ++mt)
    #pragma unroll
    for (int i = 0; i < 4; ++i) acc[mt][i] = (v8f){};

  // running stats (log2 domain)
  float mrow[MT][8], lrow[MT][8];
  #pragma unroll
  for (int mt = 0; mt < MT; ++mt)
    #pragma unroll
    for (int r = 0; r < 8; ++r) { mrow[mt][r] = -INFINITY; lrow[mt][r] = 0.0f; }

  for (int kt = 0; kt < S_LEN / KT; ++kt) {
    __syncthreads();
    // ---- cooperative K/V tile load, f32 -> f16, V transposed ----
    {
      const float* kp = K + baseKV + (size_t)kt * KT * D_DIM;
      const float* vp = V + baseKV + (size_t)kt * KT * D_DIM;
      #pragma unroll
      for (int i = 0; i < (KT * D_DIM) / 256; ++i) {
        const int idx = tid + i * 256;
        const int kk  = idx >> 6;
        const int dd  = idx & 63;
        sK [kk * LDK + dd] = (_Float16)kp[idx];
        sVt[dd * LDK + kk] = (_Float16)vp[idx];
      }
      if (kt + 1 < S_LEN / KT) {   // uniform branch: EXEC stays all-ones
        __builtin_prefetch(kp + KT * D_DIM + tid * 16, 0, 0);
        __builtin_prefetch(vp + KT * D_DIM + tid * 16, 0, 0);
      }
    }
    __syncthreads();

    // ---- per M-tile: QK^T -> online softmax -> P store (halves sc liveness) ----
    #pragma unroll
    for (int mt = 0; mt < MT; ++mt) {
      v8f sc[4];
      #pragma unroll
      for (int nt = 0; nt < 4; ++nt) {
        if (nt == 2) __builtin_amdgcn_sched_barrier(0);   // cap live B-fragments
        v16h b0, b1;
        const _Float16* krow = &sK[(nt * 16 + lc) * LDK];   // B col n = key
        #pragma unroll
        for (int v = 0; v < 8; ++v) {
          const int d0 = (v & 3) * 2 + half * 8 + (v >> 2) * 16;
          h2 p0 = *(const h2*)(krow + d0);
          h2 p1 = *(const h2*)(krow + 32 + d0);
          b0[2*v] = p0.x; b0[2*v+1] = p0.y;
          b1[2*v] = p1.x; b1[2*v+1] = p1.y;
        }
        v8f c = (v8f){};
        c = __builtin_amdgcn_wmma_f32_16x16x32_f16(false, aq[mt][0], false, b0, (short)0, c, false, false);
        c = __builtin_amdgcn_wmma_f32_16x16x32_f16(false, aq[mt][1], false, b1, (short)0, c, false, false);
        sc[nt] = c;
      }

      float alpha[8];
      #pragma unroll
      for (int r = 0; r < 8; ++r) {
        float x = red_max16(fmaxf(fmaxf(sc[0][r], sc[1][r]),
                                  fmaxf(sc[2][r], sc[3][r])));
        const float mnew = fmaxf(mrow[mt][r], x);
        alpha[r]    = __builtin_amdgcn_exp2f(mrow[mt][r] - mnew);
        mrow[mt][r] = mnew;
        float rs = 0.0f;
        #pragma unroll
        for (int nt = 0; nt < 4; ++nt) {
          const float p = __builtin_amdgcn_exp2f(sc[nt][r] - mnew);
          sc[nt][r] = p;
          rs += p;
        }
        rs = red_sum16(rs);
        lrow[mt][r] = lrow[mt][r] * alpha[r] + rs;
      }

      // P (C-layout) -> per-wave LDS; rescale accumulators
      #pragma unroll
      for (int nt = 0; nt < 4; ++nt)
        #pragma unroll
        for (int r = 0; r < 8; ++r)
          sP[wave][(mt * QT + half * 8 + r) * LDK + nt * 16 + lc] = (_Float16)sc[nt][r];

      #pragma unroll
      for (int dt = 0; dt < 4; ++dt)
        #pragma unroll
        for (int r = 0; r < 8; ++r)
          acc[mt][dt][r] *= alpha[r];
    }

    // wave-private LDS write -> read ordering
    asm volatile("s_wait_dscnt 0" ::: "memory");

    // ---- reload P as A-fragments (per M-tile) ----
    v16h ap[MT][2];
    #pragma unroll
    for (int mt = 0; mt < MT; ++mt) {
      const _Float16* prow = &sP[wave][(mt * QT + lc) * LDK];
      #pragma unroll
      for (int v = 0; v < 8; ++v) {
        const int k0 = (v & 3) * 2 + half * 8 + (v >> 2) * 16;
        h2 p0 = *(const h2*)(prow + k0);
        h2 p1 = *(const h2*)(prow + 32 + k0);
        ap[mt][0][2*v] = p0.x; ap[mt][0][2*v+1] = p0.y;
        ap[mt][1][2*v] = p1.x; ap[mt][1][2*v+1] = p1.y;
      }
    }

    // ---- acc += P V : 4 d-tiles; B-frags reused by both M-tiles ----
    #pragma unroll
    for (int dt = 0; dt < 4; ++dt) {
      if (dt == 2) __builtin_amdgcn_sched_barrier(0);   // cap live B-fragments
      v16h b0, b1;
      const _Float16* vcol = &sVt[(dt * 16 + lc) * LDK];   // B col n = dim
      #pragma unroll
      for (int v = 0; v < 8; ++v) {
        const int k0 = (v & 3) * 2 + half * 8 + (v >> 2) * 16;
        h2 p0 = *(const h2*)(vcol + k0);
        h2 p1 = *(const h2*)(vcol + 32 + k0);
        b0[2*v] = p0.x; b0[2*v+1] = p0.y;
        b1[2*v] = p1.x; b1[2*v+1] = p1.y;
      }
      #pragma unroll
      for (int mt = 0; mt < MT; ++mt) {
        acc[mt][dt] = __builtin_amdgcn_wmma_f32_16x16x32_f16(false, ap[mt][0], false, b0, (short)0, acc[mt][dt], false, false);
        acc[mt][dt] = __builtin_amdgcn_wmma_f32_16x16x32_f16(false, ap[mt][1], false, b1, (short)0, acc[mt][dt], false, false);
      }
    }
  }

  // ---- finalize: O = acc / l, write f32 ----
  #pragma unroll
  for (int mt = 0; mt < MT; ++mt)
    #pragma unroll
    for (int r = 0; r < 8; ++r) {
      const float inv = __builtin_amdgcn_rcpf(lrow[mt][r]);
      float* orow = O + baseQ + (size_t)(mt * QT + half * 8 + r) * D_DIM;
      #pragma unroll
      for (int dt = 0; dt < 4; ++dt)
        orow[dt * 16 + lc] = acc[mt][dt][r] * inv;
    }
}

extern "C" void kernel_launch(void* const* d_in, const int* in_sizes, int n_in,
                              void* d_out, int out_size, void* d_ws, size_t ws_size,
                              hipStream_t stream) {
  const float* Q = (const float*)d_in[0];
  const float* K = (const float*)d_in[1];
  const float* V = (const float*)d_in[2];
  float* Out = (float*)d_out;

  dim3 grid(S_LEN / (WAVES * MT * QT), 8 * 8);   // (8, B*H=64)
  dim3 block(256);
  flash_attn_f16_wmma<<<grid, block, 0, stream>>>(Q, K, V, Out);
}
